// CrossClipTrackingModule_51230369906753
// MI455X (gfx1250) — matrix operand order repaired and checked by reading, not compiled, not run on hardware
//
#include <hip/hip_runtime.h>

// ---------------------------------------------------------------------------
// CDNA5 (gfx1250) implementation of CrossClipTrackingModule / TrajectoryAttention
// B=2, N=3072, C=256, H=8, dh=32, F=6 (num_frames), P=512 (seq_len)
//
// All heavy GEMMs and both attention matmuls use v_wmma_f32_16x16x32_bf16
// (bf16 inputs, f32 accumulate). Working set ~80MB -> L2 resident; compute
// bound on WMMA pipe.
// ---------------------------------------------------------------------------

#define DEV __device__ __forceinline__

typedef __attribute__((ext_vector_type(16))) __bf16 v16bf;
typedef __attribute__((ext_vector_type(8)))  __bf16 bf8;
typedef __attribute__((ext_vector_type(8)))  float  v8f;

union VB { v16bf v; bf8 h[2]; };

static constexpr int   kB  = 2;
static constexpr int   kN  = 3072;
static constexpr int   kC  = 256;
static constexpr int   kH  = 8;
static constexpr int   kDH = 32;
static constexpr int   kF  = 6;
static constexpr int   kP  = 512;
static constexpr float kScale = 0.17677669529663687f;   // dh^-0.5 = 1/sqrt(32)

// D = A(16x32 bf16) x B(32x16 bf16) + C(16x16 f32)
DEV v8f wmma_bf16(v16bf a, v16bf b, v8f c) {
  return __builtin_amdgcn_wmma_f32_16x16x32_bf16(
      /*neg_a=*/false, a, /*neg_b=*/false, b,
      /*c_mod=*/(short)0, c, /*reuse_a=*/false, /*reuse_b=*/false);
}

// A-matrix fragment (16x32, 16-bit): lane holds row (lane&15);
// half = lane>>4 selects K = 16*half .. 16*half+15 contiguous.
DEV v16bf load_a_frag(const __bf16* row, int k0, int half) {
  VB u;
  u.h[0] = *(const bf8*)(row + k0 + half * 16);
  u.h[1] = *(const bf8*)(row + k0 + half * 16 + 8);
  return u.v;
}

// B-matrix fragment (32x16, 16-bit): lane holds column (lane&15);
// K = 16*half .. 16*half+15 contiguous.  `row` points at the K-run for
// this lane's column (i.e. column-major / transposed storage).
DEV v16bf load_b_frag(const __bf16* row, int k0, int half) {
  const bf8* p = (const bf8*)(row + k0 + half * 16);
  VB u; u.h[0] = p[0]; u.h[1] = p[1];
  return u.v;
}

// ---------------------------------------------------------------------------
// one-time converts
// ---------------------------------------------------------------------------
__global__ void cvt_f32_to_bf16(const float* __restrict__ src,
                                __bf16* __restrict__ dst, int n) {
  int i = blockIdx.x * blockDim.x + threadIdx.x;
  if (i < n) dst[i] = (__bf16)src[i];
}

// src [Krows, Ncols] f32 row-major  ->  dst [Ncols, Krows] bf16 row-major
__global__ void transpose_to_bf16(const float* __restrict__ src,
                                  __bf16* __restrict__ dst, int Krows, int Ncols) {
  int i = blockIdx.x * blockDim.x + threadIdx.x;
  if (i < Krows * Ncols) {
    int k = i / Ncols, n = i % Ncols;
    dst[n * Krows + k] = (__bf16)src[i];
  }
}

// ---------------------------------------------------------------------------
// GEMM 1: qkv = x @ w_qkv + b_qkv, scattered into head layouts:
//   Q,K  bf16 [B*H, N, 32]   (row-major dh)
//   Vt   bf16 [B*H, 32, N]   (transposed: contiguous over tokens)
// M=6144 K=256 Nc=768, one 16x16 tile per wave, 4 waves/block
// ---------------------------------------------------------------------------
__global__ void gemm_qkv(const __bf16* __restrict__ xb,
                         const __bf16* __restrict__ wt,     // [768,256]
                         const float*  __restrict__ bias,
                         __bf16* __restrict__ Q, __bf16* __restrict__ K,
                         __bf16* __restrict__ Vt) {
  const int wave = blockIdx.x * (blockDim.x >> 5) + (threadIdx.x >> 5);
  const int lane = threadIdx.x & 31, lm = lane & 15, half = lane >> 4;
  const int TN = (3 * kC) / 16;              // 48
  const int tm = wave / TN, tn = wave % TN;
  const __bf16* arow = xb + (tm * 16 + lm) * kC;
  const __bf16* brow = wt + (tn * 16 + lm) * kC;
  v8f acc = {};
#pragma unroll
  for (int k0 = 0; k0 < kC; k0 += 32)
    acc = wmma_bf16(load_a_frag(arow, k0, half), load_b_frag(brow, k0, half), acc);

  const int cn  = tn * 16 + lm;
  const float bv = bias[cn];
  const int sec = cn >> 8, cc = cn & 255, hh = cc >> 5, d = cc & 31;
#pragma unroll
  for (int r = 0; r < 8; ++r) {
    int t = tm * 16 + half * 8 + r;          // global token row
    int b = t / kN, nn = t % kN;
    __bf16 v = (__bf16)(acc[r] + bv);
    int hb = b * kH + hh;
    if (sec == 0)      Q [(hb * kN + nn) * kDH + d] = v;
    else if (sec == 1) K [(hb * kN + nn) * kDH + d] = v;
    else               Vt[(hb * kDH + d) * kN + nn] = v;
  }
}

// ---------------------------------------------------------------------------
// Space attention (flash-style, per frame group):
// block = (b, h, f, 32-query block), 2 waves x 16 queries.
// LDS per wave: 16x512 f32 scores (32KB) + 16x512 bf16 probs (16KB)
// ---------------------------------------------------------------------------
__global__ void space_attn(const __bf16* __restrict__ Q,
                           const __bf16* __restrict__ K,
                           const __bf16* __restrict__ Vt,
                           __bf16* __restrict__ y) {  // [B, N, F, C] bf16
  extern __shared__ char smem[];
  const int wid  = threadIdx.x >> 5;                  // 0..1
  const int lane = threadIdx.x & 31, lm = lane & 15, half = lane >> 4;
  float*  sc = (float*)(smem + wid * 49152);
  __bf16* at = (__bf16*)(smem + wid * 49152 + 32768);

  int qb = blockIdx.x % (kN / 32); int rest = blockIdx.x / (kN / 32);
  int f  = rest % kF; rest /= kF;
  int h  = rest % kH; int b = rest / kH;
  const int q0 = qb * 32 + wid * 16;
  const int bh = b * kH + h;

  const __bf16* Qb  = Q  + (bh * kN + q0) * kDH;
  const __bf16* Kb  = K  + (bh * kN + f * kP) * kDH;
  const __bf16* Vb  = Vt + bh * kDH * kN + f * kP;    // row d: Vb + d*kN

  // A fragment: 16 queries x dh=32 (single K-step)
  v16bf aq = load_a_frag(Qb + lm * kDH, 0, half);

  // ---- scores: 32 key tiles of 16, one WMMA each ----
  for (int kt = 0; kt < kP / 16; ++kt) {
    v8f z = {};
    v16bf bk = load_b_frag(Kb + (kt * 16 + lm) * kDH, 0, half);
    v8f s = wmma_bf16(aq, bk, z);
#pragma unroll
    for (int r = 0; r < 8; ++r)
      sc[(half * 8 + r) * kP + kt * 16 + lm] = s[r];
  }
  asm volatile("s_wait_dscnt 0x0" ::: "memory");

  // ---- softmax over P=512 per row: 2 lanes per row ----
  {
    float* row = sc + lm * kP + half * 256;
    float mx = -3.0e38f;
    for (int i = 0; i < 256; ++i) mx = fmaxf(mx, row[i]);
    mx = fmaxf(mx, __shfl_xor(mx, 16));
    mx *= kScale;
    float sum = 0.f;
    for (int i = 0; i < 256; ++i) {
      float p = __expf(row[i] * kScale - mx);
      row[i] = p; sum += p;
    }
    sum += __shfl_xor(sum, 16);
    float inv = 1.0f / sum;
    __bf16* arow = at + lm * kP + half * 256;
    for (int i = 0; i < 256; ++i) arow[i] = (__bf16)(row[i] * inv);
  }
  asm volatile("s_wait_dscnt 0x0" ::: "memory");

  // ---- PV: y_tile(16x32) = probs(16x512) @ V(512x32) ----
  v8f acc0 = {}, acc1 = {};
  for (int kt = 0; kt < kP / 32; ++kt) {
    v16bf ap  = load_a_frag(at + lm * kP, kt * 32, half);
    v16bf bv0 = load_b_frag(Vb + lm * kN,        kt * 32, half);
    v16bf bv1 = load_b_frag(Vb + (16 + lm) * kN, kt * 32, half);
    acc0 = wmma_bf16(ap, bv0, acc0);
    acc1 = wmma_bf16(ap, bv1, acc1);
  }
#pragma unroll
  for (int r = 0; r < 8; ++r) {
    int q = q0 + half * 8 + r;
    int base = ((b * kN + q) * kF + f) * kC + h * kDH;
    y[base + lm]      = (__bf16)acc0[r];
    y[base + 16 + lm] = (__bf16)acc1[r];
  }
}

// ---------------------------------------------------------------------------
// GEMM: kv2 = y @ w_kv + b_kv   (M=B*N*F=36864, K=256, Nc=512) -> bf16
// ---------------------------------------------------------------------------
__global__ void gemm_kv2(const __bf16* __restrict__ y,
                         const __bf16* __restrict__ wt,     // [512,256]
                         const float*  __restrict__ bias,
                         __bf16* __restrict__ kv2) {        // [B*N*F, 512]
  const int wave = blockIdx.x * (blockDim.x >> 5) + (threadIdx.x >> 5);
  const int lane = threadIdx.x & 31, lm = lane & 15, half = lane >> 4;
  const int TN = (2 * kC) / 16;              // 32
  const int tm = wave / TN, tn = wave % TN;
  const __bf16* arow = y  + (size_t)(tm * 16 + lm) * kC;
  const __bf16* brow = wt + (tn * 16 + lm) * kC;
  v8f acc = {};
#pragma unroll
  for (int k0 = 0; k0 < kC; k0 += 32)
    acc = wmma_bf16(load_a_frag(arow, k0, half), load_b_frag(brow, k0, half), acc);
  const int cn = tn * 16 + lm;
  const float bv = bias[cn];
#pragma unroll
  for (int r = 0; r < 8; ++r) {
    size_t m = (size_t)tm * 16 + half * 8 + r;
    kv2[m * (2 * kC) + cn] = (__bf16)(acc[r] + bv);
  }
}

// ---------------------------------------------------------------------------
// GEMM: q2 = (diag(y) @ w_q + b_q) * scale   (M=6144, K=256, Nc=256) -> f32
// A row for token t: y[b, nn, g=nn/P, :]
// ---------------------------------------------------------------------------
__global__ void gemm_q2(const __bf16* __restrict__ y,
                        const __bf16* __restrict__ wt,      // [256,256]
                        const float*  __restrict__ bias,
                        float* __restrict__ q2) {           // [B*N, 256]
  const int wave = blockIdx.x * (blockDim.x >> 5) + (threadIdx.x >> 5);
  const int lane = threadIdx.x & 31, lm = lane & 15, half = lane >> 4;
  const int TN = kC / 16;                    // 16
  const int tm = wave / TN, tn = wave % TN;
  int t = tm * 16 + lm;
  int b = t / kN, nn = t % kN, g = nn / kP;
  const __bf16* arow = y  + ((size_t)(b * kN + nn) * kF + g) * kC;
  const __bf16* brow = wt + (tn * 16 + lm) * kC;
  v8f acc = {};
#pragma unroll
  for (int k0 = 0; k0 < kC; k0 += 32)
    acc = wmma_bf16(load_a_frag(arow, k0, half), load_b_frag(brow, k0, half), acc);
  const int cn = tn * 16 + lm;
  const float bv = bias[cn];
#pragma unroll
  for (int r = 0; r < 8; ++r) {
    int m = tm * 16 + half * 8 + r;
    q2[(size_t)m * kC + cn] = (acc[r] + bv) * kScale;
  }
}

// ---------------------------------------------------------------------------
// Temporal attention over F=6 frame mixes (tiny, VALU):
// one thread per (token, head)
// ---------------------------------------------------------------------------
__global__ void temporal_attn(const float*  __restrict__ q2,   // [B*N, 256]
                              const __bf16* __restrict__ kv2,  // [B*N*F, 512]
                              __bf16* __restrict__ ao) {       // [B*N, 256]
  int i = blockIdx.x * blockDim.x + threadIdx.x;
  if (i >= kB * kN * kH) return;
  int hh = i & (kH - 1); int t = i >> 3;     // t = b*N + nn
  const float* qv = q2 + (size_t)t * kC + hh * kDH;
  float qreg[kDH];
#pragma unroll
  for (int d = 0; d < kDH; ++d) qreg[d] = qv[d];

  float logit[kF];
  float mx = -3.0e38f;
#pragma unroll
  for (int f = 0; f < kF; ++f) {
    const __bf16* k2 = kv2 + ((size_t)t * kF + f) * (2 * kC) + hh * kDH;
    float s = 0.f;
#pragma unroll
    for (int d = 0; d < kDH; ++d) s += qreg[d] * (float)k2[d];
    logit[f] = s;
    mx = fmaxf(mx, s);
  }
  float sum = 0.f;
#pragma unroll
  for (int f = 0; f < kF; ++f) { logit[f] = __expf(logit[f] - mx); sum += logit[f]; }
  float inv = 1.0f / sum;

  float out[kDH];
#pragma unroll
  for (int d = 0; d < kDH; ++d) out[d] = 0.f;
#pragma unroll
  for (int f = 0; f < kF; ++f) {
    float a = logit[f] * inv;
    const __bf16* v2 = kv2 + ((size_t)t * kF + f) * (2 * kC) + kC + hh * kDH;
#pragma unroll
    for (int d = 0; d < kDH; ++d) out[d] += a * (float)v2[d];
  }
  __bf16* o = ao + (size_t)t * kC + hh * kDH;
#pragma unroll
  for (int d = 0; d < kDH; ++d) o[d] = (__bf16)out[d];
}

// ---------------------------------------------------------------------------
// GEMM: out = ao @ w_proj + b_proj  (M=6144, K=256, Nc=256) -> f32 d_out
// ---------------------------------------------------------------------------
__global__ void gemm_proj(const __bf16* __restrict__ ao,
                          const __bf16* __restrict__ wt,     // [256,256]
                          const float*  __restrict__ bias,
                          float* __restrict__ out) {         // [B*N, 256] f32
  const int wave = blockIdx.x * (blockDim.x >> 5) + (threadIdx.x >> 5);
  const int lane = threadIdx.x & 31, lm = lane & 15, half = lane >> 4;
  const int TN = kC / 16;                    // 16
  const int tm = wave / TN, tn = wave % TN;
  const __bf16* arow = ao + (size_t)(tm * 16 + lm) * kC;
  const __bf16* brow = wt + (tn * 16 + lm) * kC;
  v8f acc = {};
#pragma unroll
  for (int k0 = 0; k0 < kC; k0 += 32)
    acc = wmma_bf16(load_a_frag(arow, k0, half), load_b_frag(brow, k0, half), acc);
  const int cn = tn * 16 + lm;
  const float bv = bias[cn];
#pragma unroll
  for (int r = 0; r < 8; ++r) {
    int m = tm * 16 + half * 8 + r;
    out[(size_t)m * kC + cn] = acc[r] + bv;
  }
}

// ---------------------------------------------------------------------------
// launch
// ---------------------------------------------------------------------------
extern "C" void kernel_launch(void* const* d_in, const int* in_sizes, int n_in,
                              void* d_out, int out_size, void* d_ws, size_t ws_size,
                              hipStream_t stream) {
  const float* x      = (const float*)d_in[0];
  const float* w_qkv  = (const float*)d_in[1];
  const float* b_qkv  = (const float*)d_in[2];
  const float* w_q    = (const float*)d_in[3];
  const float* b_q    = (const float*)d_in[4];
  const float* w_kv   = (const float*)d_in[5];
  const float* b_kv   = (const float*)d_in[6];
  const float* w_proj = (const float*)d_in[7];
  const float* b_proj = (const float*)d_in[8];
  float* out = (float*)d_out;

  const size_t nBNC = (size_t)kB * kN * kC;            // 1,572,864

  char* p = (char*)d_ws;
  auto alloc = [&](size_t bytes) {
    char* r = p; p += (bytes + 255) & ~(size_t)255; return r;
  };
  __bf16* xb     = (__bf16*)alloc(nBNC * 2);
  __bf16* wqkvT  = (__bf16*)alloc((size_t)3 * kC * kC * 2);
  __bf16* wqT    = (__bf16*)alloc((size_t)kC * kC * 2);
  __bf16* wkvT   = (__bf16*)alloc((size_t)2 * kC * kC * 2);
  __bf16* wprojT = (__bf16*)alloc((size_t)kC * kC * 2);
  __bf16* Qb     = (__bf16*)alloc(nBNC * 2);
  __bf16* Kb     = (__bf16*)alloc(nBNC * 2);
  __bf16* Vt     = (__bf16*)alloc(nBNC * 2);
  __bf16* y      = (__bf16*)alloc(nBNC * kF * 2);      // [B,N,F,C]
  __bf16* kv2    = (__bf16*)alloc(nBNC * kF * 2 * 2);  // [B*N*F, 2C]
  float*  q2     = (float*)alloc(nBNC * 4);
  __bf16* ao     = (__bf16*)alloc(nBNC * 2);

  // 0) converts / weight transposes
  cvt_f32_to_bf16<<<(int)((nBNC + 255) / 256), 256, 0, stream>>>(x, xb, (int)nBNC);
  transpose_to_bf16<<<(3 * kC * kC + 255) / 256, 256, 0, stream>>>(w_qkv,  wqkvT,  kC, 3 * kC);
  transpose_to_bf16<<<(kC * kC + 255) / 256,     256, 0, stream>>>(w_q,    wqT,    kC, kC);
  transpose_to_bf16<<<(2 * kC * kC + 255) / 256, 256, 0, stream>>>(w_kv,   wkvT,   kC, 2 * kC);
  transpose_to_bf16<<<(kC * kC + 255) / 256,     256, 0, stream>>>(w_proj, wprojT, kC, kC);

  // 1) qkv projection: 384 x 48 tiles, 4 waves/block
  gemm_qkv<<<(384 * 48) / 4, 128, 0, stream>>>(xb, wqkvT, b_qkv, Qb, Kb, Vt);

  // 2) space attention: B*H*F*(N/32) workgroups, 2 waves, 96KB LDS
  space_attn<<<kB * kH * kF * (kN / 32), 64, 2 * 49152, stream>>>(Qb, Kb, Vt, y);

  // 3) kv2 projection: 2304 x 32 tiles
  gemm_kv2<<<(2304 * 32) / 4, 128, 0, stream>>>(y, wkvT, b_kv, kv2);

  // 4) q2 projection (diagonal gather + scale): 384 x 16 tiles
  gemm_q2<<<(384 * 16) / 4, 128, 0, stream>>>(y, wqT, b_q, q2);

  // 5) temporal attention over F=6
  temporal_attn<<<(kB * kN * kH) / 256, 256, 0, stream>>>(q2, kv2, ao);

  // 6) output projection: 384 x 16 tiles
  gemm_proj<<<(384 * 16) / 4, 128, 0, stream>>>(ao, wprojT, b_proj, out);
}
